// QuantGELU_64269890617869
// MI455X (gfx1250) — matrix operand order repaired and checked by reading, not compiled, not run on hardware
//
#include <hip/hip_runtime.h>

// ---------------------------------------------------------------------------
// QuantGELU: out = gelu_tanh( scale[n] * (A @ w_int8[n,:].T) + bias[n] )
//   A[16384,1152] fp32, w_int8[4608,1152] int32(int8 vals), scale/bias[4608]
// Plan:
//   prepass 1: A fp32 -> bf16 (into d_ws)          [memory bound]
//   prepass 2: W int  -> bf16 exactly (|w| <= 127 is exact in bf16)
//   GEMM: bf16 WMMA 16x16x32, fp32 accum, async global->LDS tile copies
//         (ASYNCcnt-tracked), scale/bias/GELU fused epilogue.
//   Both bf16 operands (37.7 MB + 10.6 MB) fit in the 192 MB L2, so the
//   GEMM's repeated tile reads are L2 hits; HBM sees ~495 MB total vs
//   ~35 us of bf16 WMMA compute -> matrix-unit bound, as intended.
// ---------------------------------------------------------------------------

typedef __attribute__((ext_vector_type(16))) __bf16 v16bf;
typedef __attribute__((ext_vector_type(8)))  float  v8f;
typedef __attribute__((ext_vector_type(4)))  float  f4;
typedef __attribute__((ext_vector_type(4)))  int    i4;
typedef __attribute__((ext_vector_type(4)))  unsigned int u4;

#define M_DIM 16384
#define K_DIM 1152
#define N_DIM 4608

#define BM 128
#define BN 128
#define BK 32
#define KSTRIDE 48                 // 96-byte LDS rows (16B aligned chunks)
#define NTHREADS 256
#define NK (K_DIM / BK)            // 36

#if __has_builtin(__builtin_amdgcn_global_load_async_to_lds_b128)
#define ASYNC_LDS 1
#else
#define ASYNC_LDS 0
#endif

__device__ __forceinline__ void wait_asynccnt_le4() {
#if __has_builtin(__builtin_amdgcn_s_wait_asynccnt)
    __builtin_amdgcn_s_wait_asynccnt(4);
#else
    asm volatile("s_wait_asynccnt 0x4" ::: "memory");
#endif
}
__device__ __forceinline__ void wait_asynccnt_0() {
#if __has_builtin(__builtin_amdgcn_s_wait_asynccnt)
    __builtin_amdgcn_s_wait_asynccnt(0);
#else
    asm volatile("s_wait_asynccnt 0x0" ::: "memory");
#endif
}

__device__ __forceinline__ unsigned short f32_bf16(float f) {
    unsigned int u = __float_as_uint(f);
    u += 0x7FFFu + ((u >> 16) & 1u);            // RNE
    return (unsigned short)(u >> 16);
}

__device__ __forceinline__ unsigned int pack2_bf16(float lo, float hi) {
#if __has_builtin(__builtin_amdgcn_cvt_pk_bf16_f32)
    auto v = __builtin_amdgcn_cvt_pk_bf16_f32(lo, hi);
    unsigned int u;
    __builtin_memcpy(&u, &v, 4);
    return u;
#else
    return (unsigned int)f32_bf16(lo) | ((unsigned int)f32_bf16(hi) << 16);
#endif
}

__device__ __forceinline__ float gelu_tanh_f(float x) {
    const float c0 = 0.7978845608028654f;       // sqrt(2/pi)
    float u = c0 * (x + 0.044715f * x * x * x);
    float e = __expf(2.0f * u);
    float t = 1.0f - 2.0f / (e + 1.0f);         // tanh(u), overflow-safe
    return 0.5f * x * (1.0f + t);
}

// ------------------------- shared GEMM pieces ------------------------------

__device__ __forceinline__ void mma_tile(const unsigned short (*As)[KSTRIDE],
                                         const unsigned short (*Bs)[KSTRIDE],
                                         int lane, int aRowW, int bColW,
                                         v8f (&acc)[4][2]) {
    const int halfSel = lane >> 4;      // ISA 7.12.2 half-wave K split
    const int lane16  = lane & 15;
    v16bf afrag[4];
#pragma unroll
    for (int i = 0; i < 4; ++i) {
        const unsigned short* src = &As[aRowW + i * 16 + lane16][halfSel * 8];
        __builtin_memcpy((char*)&afrag[i],      src,      16);  // K k0..k0+7
        __builtin_memcpy((char*)&afrag[i] + 16, src + 16, 16);  // K k0+16..23
    }
    v16bf bfrag[2];
#pragma unroll
    for (int j = 0; j < 2; ++j) {
        const unsigned short* src = &Bs[bColW + j * 16 + lane16][halfSel * 8];
        __builtin_memcpy((char*)&bfrag[j],      src,      16);
        __builtin_memcpy((char*)&bfrag[j] + 16, src + 16, 16);
    }
#pragma unroll
    for (int i = 0; i < 4; ++i)
#pragma unroll
        for (int j = 0; j < 2; ++j)
            acc[i][j] = __builtin_amdgcn_wmma_f32_16x16x32_bf16(
                false, afrag[i], false, bfrag[j], (short)0, acc[i][j],
                false, false);
}

__device__ __forceinline__ void epilogue(v8f (&acc)[4][2], int lane,
                                         int aRowW, int bColW,
                                         int mBase, int nBase,
                                         const float* __restrict__ wscale,
                                         const float* __restrict__ bias,
                                         float* __restrict__ out) {
    const int halfSel = lane >> 4;
    const int lane16  = lane & 15;
#pragma unroll
    for (int j = 0; j < 2; ++j) {
        const int n   = nBase + bColW + j * 16 + lane16;
        const float sc = wscale[n];
        const float bn = bias[n];
#pragma unroll
        for (int i = 0; i < 4; ++i) {
            const int mRow = mBase + aRowW + i * 16 + halfSel * 8;
#pragma unroll
            for (int v = 0; v < 8; ++v) {
                float x = acc[i][j][v] * sc + bn;
                out[(long)(mRow + v) * N_DIM + n] = gelu_tanh_f(x);
            }
        }
    }
}

// ------------------------- prepass kernels ---------------------------------

__global__ __launch_bounds__(NTHREADS)
void cvt_a_bf16(const float* __restrict__ A, unsigned short* __restrict__ Ab) {
    const size_t idx = ((size_t)blockIdx.x * NTHREADS + threadIdx.x) * 8;
    f4 a0 = *(const f4*)(A + idx);
    f4 a1 = *(const f4*)(A + idx + 4);
    u4 o;
    o.x = pack2_bf16(a0.x, a0.y);
    o.y = pack2_bf16(a0.z, a0.w);
    o.z = pack2_bf16(a1.x, a1.y);
    o.w = pack2_bf16(a1.z, a1.w);
    *(u4*)(Ab + idx) = o;
}

__global__ __launch_bounds__(NTHREADS)
void cvt_w_bf16(const int* __restrict__ W, unsigned short* __restrict__ Wb) {
    const size_t idx = ((size_t)blockIdx.x * NTHREADS + threadIdx.x) * 8;
    i4 w0 = *(const i4*)(W + idx);
    i4 w1 = *(const i4*)(W + idx + 4);
    u4 o;
    o.x = pack2_bf16((float)w0.x, (float)w0.y);   // int8 is exact in bf16
    o.y = pack2_bf16((float)w0.z, (float)w0.w);
    o.z = pack2_bf16((float)w1.x, (float)w1.y);
    o.w = pack2_bf16((float)w1.z, (float)w1.w);
    *(u4*)(Wb + idx) = o;
}

// ------------------- main GEMM (pre-converted bf16) ------------------------

__global__ __launch_bounds__(NTHREADS)
void qg_gemm_pre(const unsigned short* __restrict__ Ab,
                 const unsigned short* __restrict__ Wb,
                 const float* __restrict__ wscale,
                 const float* __restrict__ bias,
                 float* __restrict__ out) {
    __shared__ unsigned short As[2][BM][KSTRIDE];
    __shared__ unsigned short Bs[2][BN][KSTRIDE];

    const int tid   = threadIdx.x;
    const int lane  = tid & 31;
    const int wave  = tid >> 5;
    const int aRowW = (wave >> 2) * 64;
    const int bColW = (wave & 3) * 32;
    const int mBase = blockIdx.y * BM;
    const int nBase = blockIdx.x * BN;

    // copy mapping: 4 threads/row (4 x 16B = 64B row), 64 rows/pass, 2 passes
    const int cpRow = tid >> 2;
    const int cpCol = (tid & 3) * 8;

    v8f acc[4][2] = {};

#if ASYNC_LDS
    auto issueTile = [&](int buf, int kt) {
#pragma unroll
        for (int p = 0; p < 2; ++p) {
            const int r = cpRow + p * 64;
            const unsigned short* ga =
                Ab + (size_t)(mBase + r) * K_DIM + (size_t)kt * BK + cpCol;
            const unsigned short* gb =
                Wb + (size_t)(nBase + r) * K_DIM + (size_t)kt * BK + cpCol;
            __builtin_amdgcn_global_load_async_to_lds_b128(
                (i4*)ga, (i4*)&As[buf][r][cpCol], 0, 0);
            __builtin_amdgcn_global_load_async_to_lds_b128(
                (i4*)gb, (i4*)&Bs[buf][r][cpCol], 0, 0);
        }
    };

    issueTile(0, 0);                                   // 4 async ops in flight
    for (int kt = 0; kt < NK; ++kt) {
        const int cur = kt & 1;
        if (kt + 1 < NK) {
            issueTile(cur ^ 1, kt + 1);                // safe: prev barrier done
            wait_asynccnt_le4();                       // tile `cur` retired
        } else {
            wait_asynccnt_0();
        }
        __syncthreads();                               // tile visible to all
        mma_tile(As[cur], Bs[cur], lane, aRowW, bColW, acc);
        __syncthreads();                               // reads done before overwrite
    }
#else
    i4 rga[2], rgb[2];
    auto loadRegs = [&](int kt) {
#pragma unroll
        for (int p = 0; p < 2; ++p) {
            const int r = cpRow + p * 64;
            rga[p] = *(const i4*)(Ab + (size_t)(mBase + r) * K_DIM +
                                  (size_t)kt * BK + cpCol);
            rgb[p] = *(const i4*)(Wb + (size_t)(nBase + r) * K_DIM +
                                  (size_t)kt * BK + cpCol);
        }
    };
    auto storeRegs = [&](int buf) {
#pragma unroll
        for (int p = 0; p < 2; ++p) {
            const int r = cpRow + p * 64;
            *(i4*)&As[buf][r][cpCol] = rga[p];
            *(i4*)&Bs[buf][r][cpCol] = rgb[p];
        }
    };
    loadRegs(0);
    storeRegs(0);
    __syncthreads();
    for (int kt = 0; kt < NK; ++kt) {
        const int cur = kt & 1;
        if (kt + 1 < NK) loadRegs(kt + 1);
        mma_tile(As[cur], Bs[cur], lane, aRowW, bColW, acc);
        if (kt + 1 < NK) {
            __syncthreads();
            storeRegs(cur ^ 1);
            __syncthreads();
        }
    }
#endif

    epilogue(acc, lane, aRowW, bColW, mBase, nBase, wscale, bias, out);
}

// ------------- fallback GEMM (dequant fused, no workspace) -----------------

__global__ __launch_bounds__(NTHREADS)
void qg_gemm_fused(const float* __restrict__ A,
                   const int*   __restrict__ W,
                   const float* __restrict__ wscale,
                   const float* __restrict__ bias,
                   float*       __restrict__ out) {
    __shared__ unsigned short As[2][BM][KSTRIDE];
    __shared__ unsigned short Bs[2][BN][KSTRIDE];

    const int tid   = threadIdx.x;
    const int lane  = tid & 31;
    const int wave  = tid >> 5;
    const int aRowW = (wave >> 2) * 64;
    const int bColW = (wave & 3) * 32;
    const int mBase = blockIdx.y * BM;
    const int nBase = blockIdx.x * BN;

    const int ldRow0 = tid >> 3;             // 0..31, 4 passes of 32 rows
    const int ldCol  = (tid & 7) << 2;       // 0,4,...,28 (floats/ints)

    f4 ra[4];
    i4 rb[4];
    auto loadTile = [&](int kt) {
        const long kOff = (long)kt * BK;
#pragma unroll
        for (int p = 0; p < 4; ++p) {
            const int r = ldRow0 + p * 32;
            ra[p] = *(const f4*)(A + (long)(mBase + r) * K_DIM + kOff + ldCol);
            rb[p] = *(const i4*)(W + (long)(nBase + r) * K_DIM + kOff + ldCol);
        }
    };
    auto storeTile = [&](int buf) {
#pragma unroll
        for (int p = 0; p < 4; ++p) {
            const int r = ldRow0 + p * 32;
            unsigned int* pa = (unsigned int*)&As[buf][r][ldCol];
            pa[0] = pack2_bf16(ra[p].x, ra[p].y);
            pa[1] = pack2_bf16(ra[p].z, ra[p].w);
            unsigned int* pb = (unsigned int*)&Bs[buf][r][ldCol];
            pb[0] = pack2_bf16((float)rb[p].x, (float)rb[p].y);
            pb[1] = pack2_bf16((float)rb[p].z, (float)rb[p].w);
        }
    };

    v8f acc[4][2] = {};
    loadTile(0);
    storeTile(0);
    __syncthreads();
    for (int kt = 0; kt < NK; ++kt) {
        const int cur = kt & 1;
        if (kt + 1 < NK) loadTile(kt + 1);
        mma_tile(As[cur], Bs[cur], lane, aRowW, bColW, acc);
        if (kt + 1 < NK) {
            __syncthreads();
            storeTile(cur ^ 1);
            __syncthreads();
        }
    }
    epilogue(acc, lane, aRowW, bColW, mBase, nBase, wscale, bias, out);
}

// ---------------------------------------------------------------------------

extern "C" void kernel_launch(void* const* d_in, const int* in_sizes, int n_in,
                              void* d_out, int out_size, void* d_ws, size_t ws_size,
                              hipStream_t stream) {
    (void)in_sizes; (void)n_in; (void)out_size;
    const float* hidden = (const float*)d_in[0];   // [8,2048,1152] fp32
    const int*   wq     = (const int*)d_in[1];     // [4608,1152] int32 (int8)
    const float* wsc    = (const float*)d_in[2];   // [4608]
    const float* bs     = (const float*)d_in[3];   // [4608]
    float* out = (float*)d_out;

    const size_t A_ELEMS = (size_t)M_DIM * K_DIM;  // 18,874,368
    const size_t W_ELEMS = (size_t)N_DIM * K_DIM;  //  5,308,416
    const size_t need    = (A_ELEMS + W_ELEMS) * sizeof(unsigned short);

    dim3 grid(N_DIM / BN, M_DIM / BM);             // 36 x 128
    dim3 block(NTHREADS);

    if (ws_size >= need) {
        unsigned short* Ab = (unsigned short*)d_ws;
        unsigned short* Wb = Ab + A_ELEMS;
        cvt_a_bf16<<<(unsigned)(A_ELEMS / 8 / NTHREADS), block, 0, stream>>>(hidden, Ab);
        cvt_w_bf16<<<(unsigned)(W_ELEMS / 8 / NTHREADS), block, 0, stream>>>(wq, Wb);
        qg_gemm_pre<<<grid, block, 0, stream>>>(Ab, Wb, wsc, bs, out);
    } else {
        qg_gemm_fused<<<grid, block, 0, stream>>>(hidden, wq, wsc, bs, out);
    }
}
